// VarPointMVSNet_82721070120991
// MI455X (gfx1250) — matrix operand order repaired and checked by reading, not compiled, not run on hardware
//
#include <hip/hip_runtime.h>
#include <hip/hip_bf16.h>

// ---------------------------------------------------------------------------
// MVSNet-style pipeline for gfx1250 (MI455X), wave32 + WMMA bf16.
//
// Input flattening assumption (jax pytree: dict keys sorted):
//   0 img_list (1,3,3,512,640) f32
//   1 cam_intrinsic (1,3,3,3)  f32
//   2 cam_extrinsic (1,3,3,4)  f32
//   3 depth_values (64,)       f32
//   4..14  img_params biases  order b0,b1,b10,b2..b9
//   15..25 img_params weights order w0,w1,w10,w2..w9
//   26..47 vol_params sorted: db0,db1,db2,dw0,dw1,dw2,fb,fw,
//          lb0..lb3,lw0..lw3,ub0..ub2,uw0..uw2
// ---------------------------------------------------------------------------

typedef __attribute__((ext_vector_type(16))) __bf16 v16bf;
typedef __attribute__((ext_vector_type(8)))  float  v8f;

__device__ __forceinline__ __bf16 f2bf(float f) {
  // round-to-nearest-even f32 -> bf16, pure bit manipulation (no __bf16 arith)
  unsigned int u = __builtin_bit_cast(unsigned int, f);
  unsigned int r = (u + 0x7FFFu + ((u >> 16) & 1u)) >> 16;
  unsigned short s = (unsigned short)r;
  return __builtin_bit_cast(__bf16, s);
}

// --- WMMA fragment gathers per CDNA5 ISA 16-bit layouts (wave32) -----------
// A: 16x32 bf16, row m = lane&15, K pairs per VGPR; lanes>=16 get K offset +8.
__device__ __forceinline__ v16bf load_a_frag(const __bf16* sA, int lane) {
  v16bf a;
  int m = lane & 15, hf = lane >> 4;
#pragma unroll
  for (int j = 0; j < 8; ++j) {
    int kb = (j < 4 ? 2 * j : 16 + 2 * (j - 4)) + 8 * hf;
    a[2 * j]     = sA[m * 32 + kb];
    a[2 * j + 1] = sA[m * 32 + kb + 1];
  }
  return a;
}
// B: 32x16 bf16 subtile of a [32k][32n] LDS tile starting at column ncol0.
// col n = lane&15; VGPR j holds rows K=2j (j<4) or 16+2(j-4) (j>=4), +8 for hi lanes, and K+1.
__device__ __forceinline__ v16bf load_b_frag32(const __bf16* sB, int lane,
                                               int ncol0) {
  v16bf b;
  int n = (lane & 15) + ncol0, hf = lane >> 4;
#pragma unroll
  for (int j = 0; j < 8; ++j) {
    int k = (j < 4 ? 2 * j : 16 + 2 * (j - 4)) + 8 * hf;
    b[2 * j]     = sB[k * 32 + n];
    b[2 * j + 1] = sB[(k + 1) * 32 + n];
  }
  return b;
}

// ---------------------------------------------------------------------------
// Implicit-GEMM conv2d (NCHW, SAME). One wave per 16(outch) x 32(pixel) tile;
// the A (weight) fragment is reused across two back-to-back WMMAs.
// ---------------------------------------------------------------------------
__global__ void conv2d_wmma(const float* __restrict__ in,
                            const float* __restrict__ wgt,
                            const float* __restrict__ bias,
                            float* __restrict__ out,
                            int Cin, int Cout, int Hin, int Win,
                            int Hout, int Wout, int kk, int stride, int pad,
                            int relu) {
  __shared__ __bf16 sA[16 * 32];
  __shared__ __bf16 sB[32 * 32];
  const int lane = threadIdx.x;
  const int spat = Hout * Wout;
  const int n0 = blockIdx.x * 32;
  const int m0 = blockIdx.y * 16;
  const float* inI = in + (size_t)blockIdx.z * Cin * Hin * Win;
  float* outI = out + (size_t)blockIdx.z * Cout * spat;
  const int Ktot = Cin * kk * kk;

  __builtin_amdgcn_s_wait_tensorcnt(0);

  v8f acc0 = {}, acc1 = {};
  for (int k0 = 0; k0 < Ktot; k0 += 32) {
    for (int t = lane; t < 512; t += 32) {           // A: weights [16m][32k]
      int m = t >> 5, k = t & 31;
      int gm = m0 + m, gk = k0 + k;
      float v = 0.f;
      if (gm < Cout && gk < Ktot) v = wgt[(size_t)gm * Ktot + gk];
      sA[t] = f2bf(v);
    }
    for (int t = lane; t < 1024; t += 32) {          // B: im2col [32k][32n]
      int k = t >> 5, n = t & 31;
      int gk = k0 + k, gn = n0 + n;
      float v = 0.f;
      if (gk < Ktot && gn < spat) {
        int ci = gk / (kk * kk), r = gk % (kk * kk);
        int ky = r / kk, kx = r % kk;
        int oy = gn / Wout, ox = gn % Wout;
        int iy = oy * stride + ky - pad;
        int ix = ox * stride + kx - pad;
        if (iy >= 0 && iy < Hin && ix >= 0 && ix < Win)
          v = inI[((size_t)ci * Hin + iy) * Win + ix];
      }
      sB[t] = f2bf(v);
    }
    __syncthreads();
    v16bf a  = load_a_frag(sA, lane);
    v16bf b0 = load_b_frag32(sB, lane, 0);
    v16bf b1 = load_b_frag32(sB, lane, 16);
    acc0 = __builtin_amdgcn_wmma_f32_16x16x32_bf16(false, a, false, b0,
                                                   (short)0, acc0, false, false);
    acc1 = __builtin_amdgcn_wmma_f32_16x16x32_bf16(false, a, false, b1,
                                                   (short)0, acc1, false, false);
    __syncthreads();
  }
  int hf = lane >> 4, n = lane & 15;
#pragma unroll
  for (int nt = 0; nt < 2; ++nt) {
    int gn = n0 + nt * 16 + n;
    if (gn < spat) {
      const v8f& acc = nt ? acc1 : acc0;
#pragma unroll
      for (int r = 0; r < 8; ++r) {
        int gm = m0 + r + 8 * hf;
        if (gm < Cout) {
          float v = acc[r] + bias[gm];
          if (relu) v = fmaxf(v, 0.f);
          outI[(size_t)gm * spat + gn] = v;
        }
      }
    }
  }
}

// ---------------------------------------------------------------------------
// Implicit-GEMM conv3d (NCDHW, SAME, k=3). Same 16x32 tiling scheme.
// ---------------------------------------------------------------------------
__global__ void conv3d_wmma(const float* __restrict__ in,
                            const float* __restrict__ wgt,
                            const float* __restrict__ bias,
                            float* __restrict__ out,
                            int Cin, int Cout, int Di, int Hi, int Wi,
                            int Do_, int Ho_, int Wo_, int stride, int pad,
                            int relu) {
  __shared__ __bf16 sA[16 * 32];
  __shared__ __bf16 sB[32 * 32];
  const int lane = threadIdx.x;
  const int spat = Do_ * Ho_ * Wo_;
  const int n0 = blockIdx.x * 32;
  const int m0 = blockIdx.y * 16;
  const int Ktot = Cin * 27;

  v8f acc0 = {}, acc1 = {};
  for (int k0 = 0; k0 < Ktot; k0 += 32) {
    for (int t = lane; t < 512; t += 32) {
      int m = t >> 5, k = t & 31;
      int gm = m0 + m, gk = k0 + k;
      float v = 0.f;
      if (gm < Cout && gk < Ktot) v = wgt[(size_t)gm * Ktot + gk];
      sA[t] = f2bf(v);
    }
    for (int t = lane; t < 1024; t += 32) {
      int k = t >> 5, n = t & 31;
      int gk = k0 + k, gn = n0 + n;
      float v = 0.f;
      if (gk < Ktot && gn < spat) {
        int ci = gk / 27, r = gk % 27;
        int kd = r / 9, kh = (r % 9) / 3, kw = r % 3;
        int od = gn / (Ho_ * Wo_), rem = gn % (Ho_ * Wo_);
        int oh = rem / Wo_, ow = rem % Wo_;
        int id = od * stride + kd - pad;
        int ih = oh * stride + kh - pad;
        int iw = ow * stride + kw - pad;
        if (id >= 0 && id < Di && ih >= 0 && ih < Hi && iw >= 0 && iw < Wi)
          v = in[((size_t)ci * Di + id) * Hi * Wi + (size_t)ih * Wi + iw];
      }
      sB[t] = f2bf(v);
    }
    __syncthreads();
    v16bf a  = load_a_frag(sA, lane);
    v16bf b0 = load_b_frag32(sB, lane, 0);
    v16bf b1 = load_b_frag32(sB, lane, 16);
    acc0 = __builtin_amdgcn_wmma_f32_16x16x32_bf16(false, a, false, b0,
                                                   (short)0, acc0, false, false);
    acc1 = __builtin_amdgcn_wmma_f32_16x16x32_bf16(false, a, false, b1,
                                                   (short)0, acc1, false, false);
    __syncthreads();
  }
  int hf = lane >> 4, n = lane & 15;
#pragma unroll
  for (int nt = 0; nt < 2; ++nt) {
    int gn = n0 + nt * 16 + n;
    if (gn < spat) {
      const v8f& acc = nt ? acc1 : acc0;
#pragma unroll
      for (int r = 0; r < 8; ++r) {
        int gm = m0 + r + 8 * hf;
        if (gm < Cout) {
          float v = acc[r] + bias[gm];
          if (relu) v = fmaxf(v, 0.f);
          out[(size_t)gm * spat + gn] = v;
        }
      }
    }
  }
}

// ---------------------------------------------------------------------------
// conv_transpose3d stride 2 SAME (gather form), fused relu + skip-add.
// out = relu(deconv(in)+bias) + skip
// ---------------------------------------------------------------------------
__global__ void deconv3d_add(const float* __restrict__ in,
                             const float* __restrict__ wgt,
                             const float* __restrict__ bias,
                             const float* __restrict__ skip,
                             float* __restrict__ out,
                             int Cin, int Cout, int Di, int Hi, int Wi,
                             int Do_, int Ho_, int Wo_) {
  size_t idx = (size_t)blockIdx.x * blockDim.x + threadIdx.x;
  size_t tot = (size_t)Cout * Do_ * Ho_ * Wo_;
  if (idx >= tot) return;
  int w = (int)(idx % Wo_);
  int h = (int)((idx / Wo_) % Ho_);
  int d = (int)((idx / ((size_t)Wo_ * Ho_)) % Do_);
  int o = (int)(idx / ((size_t)Wo_ * Ho_ * Do_));
  float acc = bias[o];
  for (int kd = 0; kd < 3; ++kd) {
    int td = d + 1 - kd;
    if (td < 0 || (td & 1)) continue;
    int id = td >> 1; if (id >= Di) continue;
    for (int kh = 0; kh < 3; ++kh) {
      int th = h + 1 - kh;
      if (th < 0 || (th & 1)) continue;
      int ih = th >> 1; if (ih >= Hi) continue;
      for (int kw = 0; kw < 3; ++kw) {
        int tw = w + 1 - kw;
        if (tw < 0 || (tw & 1)) continue;
        int iw = tw >> 1; if (iw >= Wi) continue;
        size_t sp = ((size_t)id * Hi + ih) * Wi + iw;
        size_t sstride = (size_t)Di * Hi * Wi;
        int wk = kd * 9 + kh * 3 + kw;
        for (int ci = 0; ci < Cin; ++ci)
          acc += wgt[((size_t)o * Cin + ci) * 27 + wk] * in[(size_t)ci * sstride + sp];
      }
    }
  }
  out[idx] = fmaxf(acc, 0.f) + skip[idx];
}

// ---------------------------------------------------------------------------
// Camera precompute (1 thread): Kh per view, inv(Kh0), inv(R0), t0, R/t per v.
// cw layout: [0..8]=Kinv [9..17]=R0inv [18..20]=t0 then per-view 21 floats:
//   R(9) t(3) Kh(9) at 21+v*21
// ---------------------------------------------------------------------------
__device__ __forceinline__ void inv3(const float* a, float* o) {
  float det = a[0] * (a[4] * a[8] - a[5] * a[7]) -
              a[1] * (a[3] * a[8] - a[5] * a[6]) +
              a[2] * (a[3] * a[7] - a[4] * a[6]);
  float id = 1.f / det;
  o[0] = (a[4] * a[8] - a[5] * a[7]) * id;
  o[1] = (a[2] * a[7] - a[1] * a[8]) * id;
  o[2] = (a[1] * a[5] - a[2] * a[4]) * id;
  o[3] = (a[5] * a[6] - a[3] * a[8]) * id;
  o[4] = (a[0] * a[8] - a[2] * a[6]) * id;
  o[5] = (a[2] * a[3] - a[0] * a[5]) * id;
  o[6] = (a[3] * a[7] - a[4] * a[6]) * id;
  o[7] = (a[1] * a[6] - a[0] * a[7]) * id;
  o[8] = (a[0] * a[4] - a[1] * a[3]) * id;
}

__global__ void cam_setup(const float* __restrict__ K,
                          const float* __restrict__ E,
                          float* __restrict__ cw) {
  if (threadIdx.x != 0 || blockIdx.x != 0) return;
  float Kh[3][9];
  for (int v = 0; v < 3; ++v)
    for (int i = 0; i < 3; ++i)
      for (int j = 0; j < 3; ++j)
        Kh[v][i * 3 + j] = K[v * 9 + i * 3 + j] * (i < 2 ? 0.5f : 1.f);
  inv3(Kh[0], cw + 0);
  float R0[9];
  for (int i = 0; i < 3; ++i) {
    for (int j = 0; j < 3; ++j) R0[i * 3 + j] = E[i * 4 + j];
    cw[18 + i] = E[i * 4 + 3];
  }
  inv3(R0, cw + 9);
  for (int v = 0; v < 3; ++v) {
    float* p = cw + 21 + v * 21;
    for (int i = 0; i < 3; ++i) {
      for (int j = 0; j < 3; ++j) p[i * 3 + j] = E[(v * 3 + i) * 4 + j];
      p[9 + i] = E[(v * 3 + i) * 4 + 3];
    }
    for (int i = 0; i < 9; ++i) p[12 + i] = Kh[v][i];
  }
}

__device__ __forceinline__ void mat3v(const float* m, float x, float y, float z,
                                      float* o) {
  o[0] = m[0] * x + m[1] * y + m[2] * z;
  o[1] = m[3] * x + m[4] * y + m[5] * z;
  o[2] = m[6] * x + m[7] * y + m[8] * z;
}

// ---------------------------------------------------------------------------
// Cost volume: per (d,pixel) point, project into 3 views, bilinear-sample
// C=64 features, variance across views. featL: (3,64,64,80) f32.
// ---------------------------------------------------------------------------
__global__ void cost_volume_k(const float* __restrict__ featL,
                              const float* __restrict__ dvals,
                              const float* __restrict__ cw,
                              float* __restrict__ cost) {
  int n = blockIdx.x * blockDim.x + threadIdx.x;
  if (n >= 64 * 64 * 80) return;
  int pix = n % 5120, d = n / 5120;
  float px = (float)(pix % 80) + 0.5f;
  float py = (float)(pix / 80) + 0.5f;
  float uv[3];
  mat3v(cw, px, py, 1.f, uv);
  float depth = dvals[d];
  float c0[3] = {uv[0] * depth - cw[18], uv[1] * depth - cw[19],
                 uv[2] * depth - cw[20]};
  float wp[3];
  mat3v(cw + 9, c0[0], c0[1], c0[2], wp);

  float wtb[3][4];
  int idx4[3][4];
  float zmask[3];
  for (int v = 0; v < 3; ++v) {
    const float* p = cw + 21 + v * 21;
    float cam[3];
    mat3v(p, wp[0], wp[1], wp[2], cam);
    cam[0] += p[9]; cam[1] += p[10]; cam[2] += p[11];
    float pv[3];
    mat3v(p + 12, cam[0], cam[1], cam[2], pv);
    float z = pv[2];
    float zs = (fabsf(z) < 1e-6f) ? 1e-6f : z;
    float x = pv[0] / zs, y = pv[1] / zs;
    float x0 = floorf(x), y0 = floorf(y);
    float a = x - x0, bb = y - y0;
    int x0i = min(max((int)x0, 0), 79), x1i = min(x0i + 1, 79);
    int y0i = min(max((int)y0, 0), 63), y1i = min(y0i + 1, 63);
    wtb[v][0] = (1.f - a) * (1.f - bb);
    wtb[v][1] = a * (1.f - bb);
    wtb[v][2] = (1.f - a) * bb;
    wtb[v][3] = a * bb;
    idx4[v][0] = y0i * 80 + x0i;
    idx4[v][1] = y0i * 80 + x1i;
    idx4[v][2] = y1i * 80 + x0i;
    idx4[v][3] = y1i * 80 + x1i;
    zmask[v] = (z > 0.f) ? 1.f : 0.f;
    __builtin_prefetch(featL + (size_t)v * 64 * 5120 + idx4[v][0], 0, 1);
  }
  for (int c = 0; c < 64; ++c) {
    float s = 0.f, s2 = 0.f;
    for (int v = 0; v < 3; ++v) {
      const float* f = featL + ((size_t)v * 64 + c) * 5120;
      float val = f[idx4[v][0]] * wtb[v][0] + f[idx4[v][1]] * wtb[v][1] +
                  f[idx4[v][2]] * wtb[v][2] + f[idx4[v][3]] * wtb[v][3];
      val *= zmask[v];
      s += val;
      s2 += val * val;
    }
    float m = s * (1.f / 3.f);
    cost[(size_t)c * 327680 + n] = s2 * (1.f / 3.f) - m * m;
  }
}

// ---------------------------------------------------------------------------
// softmax over depth of -vol[0], expected depth.
// ---------------------------------------------------------------------------
__global__ void depth_regress(const float* __restrict__ vol,
                              const float* __restrict__ dvals,
                              float* __restrict__ out) {
  int n = blockIdx.x * blockDim.x + threadIdx.x;
  if (n >= 5120) return;
  float mx = -1e30f;
  for (int d = 0; d < 64; ++d) mx = fmaxf(mx, -vol[(size_t)d * 5120 + n]);
  float s = 0.f, sd = 0.f;
  for (int d = 0; d < 64; ++d) {
    float e = __expf(-vol[(size_t)d * 5120 + n] - mx);
    s += e;
    sd += e * dvals[d];
  }
  out[n] = sd / s;
}

// ---------------------------------------------------------------------------
extern "C" void kernel_launch(void* const* d_in, const int* in_sizes, int n_in,
                              void* d_out, int out_size, void* d_ws,
                              size_t ws_size, hipStream_t stream) {
  (void)in_sizes; (void)n_in; (void)out_size; (void)ws_size;
  const float* img = (const float*)d_in[0];
  const float* Kin = (const float*)d_in[1];
  const float* Ext = (const float*)d_in[2];
  const float* dv  = (const float*)d_in[3];

  // sorted-key position of img layer i among {0,1,10,2..9}
  static const int pos[11] = {0, 1, 3, 4, 5, 6, 7, 8, 9, 10, 2};

  float* ws = (float*)d_ws;
  const size_t BUF = 8u * 1024u * 1024u;  // 8M floats per ping-pong buffer
  float* bufA  = ws;
  float* bufB  = ws + BUF;
  float* featL = ws + 2 * BUF;                 // 3*64*64*80 = 983040
  float* camw  = featL + 1000000;              // 84 floats used (256 reserved)
  float* cost  = camw + 256;                   // 64*327680
  float* c0b   = cost + 20971520;              // 8*327680
  float* d1b   = c0b + 2621440;                // 16*40960
  float* c1b   = d1b + 655360;                 // 16*40960
  float* d2b   = c1b + 655360;                 // 32*5120
  float* c2b   = d2b + 163840;                 // 32*5120
  float* d3b   = c2b + 163840;                 // 64*640
  float* c3b   = d3b + 40960;                  // 64*640
  float* u2b   = c3b + 40960;                  // 32*5120
  float* u1b   = u2b + 163840;                 // 16*40960
  float* u0b   = u1b + 655360;                 // 8*327680
  float* volb  = u0b + 2621440;                // 1*327680

  // ---- image feature pyramid ----
  static const int ci_[11] = {3, 8, 8, 16, 16, 16, 32, 32, 32, 64, 64};
  static const int co_[11] = {8, 8, 16, 16, 16, 32, 32, 32, 64, 64, 64};
  static const int kk_[11] = {3, 3, 5, 3, 3, 5, 3, 3, 5, 3, 3};
  static const int st_[11] = {1, 1, 2, 1, 1, 2, 1, 1, 2, 1, 1};
  const float* cur = img;
  int Hc = 512, Wc = 640;
  for (int i = 0; i < 11; ++i) {
    int Ho = (st_[i] == 2) ? Hc / 2 : Hc;
    int Wo = (st_[i] == 2) ? Wc / 2 : Wc;
    float* dst = (i == 10) ? featL : ((i & 1) ? bufB : bufA);
    const float* wgt = (const float*)d_in[15 + pos[i]];
    const float* bia = (const float*)d_in[4 + pos[i]];
    int spat = Ho * Wo;
    dim3 g((spat + 31) / 32, (co_[i] + 15) / 16, 3);
    conv2d_wmma<<<g, 32, 0, stream>>>(cur, wgt, bia, dst, ci_[i], co_[i], Hc,
                                      Wc, Ho, Wo, kk_[i], st_[i], 1,
                                      (i < 10) ? 1 : 0);
    cur = dst;
    Hc = Ho; Wc = Wo;
  }

  // ---- cost volume ----
  cam_setup<<<1, 1, 0, stream>>>(Kin, Ext, camw);
  cost_volume_k<<<(327680 + 127) / 128, 128, 0, stream>>>(featL, dv, camw, cost);

  // ---- 3D U-Net ----
  const int db0=26, db1=27, db2=28, dw0=29, dw1=30, dw2=31, fb=32, fwi=33;
  const int lb0=34, lb1=35, lb2=36, lb3=37, lw0=38, lw1=39, lw2=40, lw3=41;
  const int ub0=42, ub1=43, ub2=44, uw0=45, uw1=46, uw2=47;

  auto conv3 = [&](const float* in_, int widx, int bidx, float* out_, int Cin,
                   int Cout, int Di, int Hi, int Wi, int s, int pad, int relu) {
    int Do_ = Di / s, Ho_ = Hi / s, Wo_ = Wi / s;
    int spat = Do_ * Ho_ * Wo_;
    dim3 g((spat + 31) / 32, (Cout + 15) / 16, 1);
    conv3d_wmma<<<g, 32, 0, stream>>>(in_, (const float*)d_in[widx],
                                      (const float*)d_in[bidx], out_, Cin, Cout,
                                      Di, Hi, Wi, Do_, Ho_, Wo_, s, pad, relu);
  };
  auto deconv = [&](const float* in_, int widx, int bidx, const float* skip,
                    float* out_, int Cin, int Cout, int Di, int Hi, int Wi) {
    int Do_ = Di * 2, Ho_ = Hi * 2, Wo_ = Wi * 2;
    size_t tot = (size_t)Cout * Do_ * Ho_ * Wo_;
    deconv3d_add<<<(unsigned)((tot + 255) / 256), 256, 0, stream>>>(
        in_, (const float*)d_in[widx], (const float*)d_in[bidx], skip, out_,
        Cin, Cout, Di, Hi, Wi, Do_, Ho_, Wo_);
  };

  // SAME padding: k=3 s=1 -> pad_before 1 ; k=3 s=2 (even in) -> pad_before 0
  conv3(cost, lw0, lb0, c0b, 64,  8, 64, 64, 80, 1, 1, 1);   // c0
  conv3(cost, dw0, db0, d1b, 64, 16, 64, 64, 80, 2, 0, 1);   // d1
  conv3(d1b,  lw1, lb1, c1b, 16, 16, 32, 32, 40, 1, 1, 1);   // c1
  conv3(d1b,  dw1, db1, d2b, 16, 32, 32, 32, 40, 2, 0, 1);   // d2
  conv3(d2b,  lw2, lb2, c2b, 32, 32, 16, 16, 20, 1, 1, 1);   // c2
  conv3(d2b,  dw2, db2, d3b, 32, 64, 16, 16, 20, 2, 0, 1);   // d3
  conv3(d3b,  lw3, lb3, c3b, 64, 64,  8,  8, 10, 1, 1, 1);   // c3
  deconv(c3b, uw0, ub0, c2b, u2b, 64, 32,  8,  8, 10);       // u2
  deconv(u2b, uw1, ub1, c1b, u1b, 32, 16, 16, 16, 20);       // u1
  deconv(u1b, uw2, ub2, c0b, u0b, 16,  8, 32, 32, 40);       // u0
  conv3(u0b,  fwi, fb,  volb, 8,  1, 64, 64, 80, 1, 1, 0);   // vol

  // ---- softmax depth regression ----
  depth_regress<<<(5120 + 255) / 256, 256, 0, stream>>>(volb, dv, (float*)d_out);
}
